// SMCSWT_56092272886052
// MI455X (gfx1250) — matrix operand order repaired and checked by reading, not compiled, not run on hardware
//
#include <hip/hip_runtime.h>
#include <hip/hip_bf16.h>

typedef __attribute__((ext_vector_type(16))) __bf16 v16bf;
typedef __attribute__((ext_vector_type(8)))  float  v8f;

// Try the CDNA5 async LDS-DMA path (global_load_async_to_lds_b128) via inline
// asm for the attention K-tile staging.  Set to 0 to fall back to VGPR copies.
#define USE_ASYNC_LDS 1

union BFrag {
    v16bf v;
    int4 q[2];
    __bf16 b[16];
    unsigned short u[16];
};

// f32 -> bf16 through the native fptrunc: on gfx1250 this should lower to
// v_cvt_pk_bf16_f32 (1 VALU op) instead of the 3-op software RNE sequence.
__device__ __forceinline__ unsigned short f2bf(float f) {
    __bf16 h = (__bf16)f;
    unsigned short u;
    __builtin_memcpy(&u, &h, 2);
    return u;
}
__device__ __forceinline__ float bf2f(unsigned short h) {
    return __uint_as_float(((unsigned int)h) << 16);
}

// ---------------------------------------------------------------------------
// K1: convert qk_w|v_w to bf16 (concat rows: 0..511 = qk, 512..767 = v) and
// build the shuffle-folded, bf16 out-projection weight:
//   Wp[j][c] = proj_w[j][(c&63)*4 + (c>>6)]
// ---------------------------------------------------------------------------
__global__ __launch_bounds__(256) void prep_weights(
        const float* __restrict__ qk_w, const float* __restrict__ v_w,
        const float* __restrict__ proj_w,
        unsigned short* __restrict__ wqkv, unsigned short* __restrict__ wp) {
    int tid = blockIdx.x * 256 + threadIdx.x;      // 0 .. 262143
    if (tid < 768 * 256) {
        int j = tid >> 8, c = tid & 255;
        float val = (j < 512) ? qk_w[(size_t)j * 256 + c]
                              : v_w[(size_t)(j - 512) * 256 + c];
        wqkv[tid] = f2bf(val);
    } else {
        int t2 = tid - 768 * 256;                  // 0 .. 65535
        int j = t2 >> 8, c = t2 & 255;
        int src = (c & 63) * 4 + (c >> 6);         // inverse channel shuffle
        wp[t2] = f2bf(proj_w[(size_t)j * 256 + src]);
    }
}

// ---------------------------------------------------------------------------
// K2: qkv = x @ [qk_w; v_w]^T   (M=131072, N=768, K=256), bf16 out.
// Block = 128 rows x 64 cols, 8 waves; wave r owns a 16-row band, 4 N-tiles.
// A converted f32->bf16 in-register; 8 k-steps of 32.
// ---------------------------------------------------------------------------
__global__ __launch_bounds__(256) void qkv_gemm(
        const float* __restrict__ x, const unsigned short* __restrict__ wqkv,
        unsigned short* __restrict__ qkv) {
    int wave = threadIdx.x >> 5, lane = threadIdx.x & 31;
    int half = lane >> 4, l16 = lane & 15;
    int M0 = blockIdx.x * 128, N0 = blockIdx.y * 64;

    int arow = M0 + wave * 16 + l16;               // A-fragment row (m = l%16)
    const float* xrow = x + (size_t)arow * 256;

    v8f acc[4] = {};
    #pragma unroll
    for (int ks = 0; ks < 8; ks++) {
        int k0 = ks * 32;
        if (ks < 7) __builtin_prefetch(xrow + k0 + 32, 0, 0);
        BFrag a;
        float4 f0 = *(const float4*)(xrow + k0 + half * 16);
        float4 f1 = *(const float4*)(xrow + k0 + half * 16 + 4);
        float4 f2 = *(const float4*)(xrow + k0 + half * 16 + 8);
        float4 f3 = *(const float4*)(xrow + k0 + half * 16 + 12);
        a.b[0]  = (__bf16)f0.x; a.b[1]  = (__bf16)f0.y;
        a.b[2]  = (__bf16)f0.z; a.b[3]  = (__bf16)f0.w;
        a.b[4]  = (__bf16)f1.x; a.b[5]  = (__bf16)f1.y;
        a.b[6]  = (__bf16)f1.z; a.b[7]  = (__bf16)f1.w;
        a.b[8]  = (__bf16)f2.x; a.b[9]  = (__bf16)f2.y;
        a.b[10] = (__bf16)f2.z; a.b[11] = (__bf16)f2.w;
        a.b[12] = (__bf16)f3.x; a.b[13] = (__bf16)f3.y;
        a.b[14] = (__bf16)f3.z; a.b[15] = (__bf16)f3.w;
        #pragma unroll
        for (int nt = 0; nt < 4; nt++) {
            BFrag bw;   // B[k][n] = W[n][k]: lane = row n, contiguous k
            const int4* bp = (const int4*)(wqkv + (size_t)(N0 + nt * 16 + l16) * 256
                                                 + k0 + half * 16);
            bw.q[0] = bp[0]; bw.q[1] = bp[1];
            acc[nt] = __builtin_amdgcn_wmma_f32_16x16x32_bf16(
                false, a.v, false, bw.v, (short)0, acc[nt], false, false);
        }
    }
    #pragma unroll
    for (int nt = 0; nt < 4; nt++)
        #pragma unroll
        for (int re = 0; re < 8; re++) {
            int r_out = M0 + wave * 16 + half * 8 + re;   // D layout rows
            int c_out = N0 + nt * 16 + l16;
            qkv[(size_t)r_out * 768 + c_out] = f2bf(acc[nt][re]);
        }
}

// ---------------------------------------------------------------------------
// K3: per (batch, branch, window, head) attention + fused LEPE depthwise conv.
// Window = 128 tokens, hd = 32.  S = QK^T: one WMMA per 16x16 tile (K=32).
// K tile staged with global_load_async_to_lds_b128 (ASYNCcnt path); softmax
// in-register (shfl_xor over 16-lane halves); P -> LDS; PV WMMAs; LEPE 3x3
// conv read from the V^T LDS tile; write y (bf16, concat layout).
// ---------------------------------------------------------------------------
__global__ __launch_bounds__(256) void attn_kernel(
        const unsigned short* __restrict__ qkv,
        const float* __restrict__ cw0, const float* __restrict__ cb0,
        const float* __restrict__ cw1, const float* __restrict__ cb1,
        unsigned short* __restrict__ y) {
    __shared__ __align__(16) unsigned short Ksh[128 * 40];   // K rows, pad 40
    __shared__ __align__(16) unsigned short Vt[32 * 136];    // V^T, pad 136
    __shared__ __align__(16) unsigned short Psh[128 * 136];  // P, pad 136

    int bid = blockIdx.x;
    int b = bid >> 6;
    int branch = (bid >> 5) & 1;
    int w0 = (bid >> 2) & 7;
    int h4 = bid & 3;
    int cb = branch * 128 + h4 * 32;               // channel base within 256
    int tid = threadIdx.x;

    auto tok2n = [&](int t) -> int {
        return branch == 0 ? ((t >> 2) * 32 + w0 * 4 + (t & 3)) : (w0 * 128 + t);
    };

    {   // stage K rows: 128 tokens x 32 dims bf16 (32B per thread)
        int t = tid >> 1, hk = tid & 1;
        size_t src = ((size_t)(b * 1024 + tok2n(t))) * 768 + 256 + cb + hk * 16;
#if USE_ASYNC_LDS
        // ISA: INST_OFFSET is added to BOTH the LDS and the global address,
        // so one VDST/VADDR pair covers both 16B halves.
        unsigned lds0 = (unsigned)(uintptr_t)(Ksh + t * 40 + hk * 16);
        unsigned long long g0 = (unsigned long long)(uintptr_t)(qkv + src);
        asm volatile(
            "global_load_async_to_lds_b128 %0, %1, off\n"
            "global_load_async_to_lds_b128 %0, %1, off offset:16"
            :
            : "v"(lds0), "v"(g0)
            : "memory");
#else
        const int4* p = (const int4*)(qkv + src);
        int4* dp = (int4*)(Ksh + t * 40 + hk * 16);
        dp[0] = p[0]; dp[1] = p[1];
#endif
    }
    {   // stage V transposed: Vt[d][t]
        int t = tid >> 1, hk = tid & 1;
        size_t src = ((size_t)(b * 1024 + tok2n(t))) * 768 + 512 + cb + hk * 16;
        BFrag tmp;
        const int4* p = (const int4*)(qkv + src);
        tmp.q[0] = p[0]; tmp.q[1] = p[1];
        #pragma unroll
        for (int i = 0; i < 16; i++) Vt[(hk * 16 + i) * 136 + t] = tmp.u[i];
    }
#if USE_ASYNC_LDS
    asm volatile("s_wait_asynccnt 0x0" ::: "memory");
#endif
    __syncthreads();

    int wave = tid >> 5, lane = tid & 31, half = lane >> 4, l16 = lane & 15;

    BFrag qa;   // Q A-fragment for this wave's 16-row band, from global
    {
        int t = wave * 16 + l16;
        size_t qr = ((size_t)(b * 1024 + tok2n(t))) * 768 + cb;
        qa.q[0] = *(const int4*)(qkv + qr + half * 16);
        qa.q[1] = *(const int4*)(qkv + qr + half * 16 + 8);
    }

    v8f s[8];
    #pragma unroll
    for (int c = 0; c < 8; c++) {       // S band: 8 column tiles, K=hd=32
        BFrag kb;
        const int4* kp = (const int4*)(Ksh + (c * 16 + l16) * 40 + half * 16);
        kb.q[0] = kp[0]; kb.q[1] = kp[1];
        v8f z = {};
        s[c] = __builtin_amdgcn_wmma_f32_16x16x32_bf16(
            false, qa.v, false, kb.v, (short)0, z, false, false);
    }

    const float scale = 0.17677669529663687f;      // 32^-0.5
    float mx[8], sm[8], inv[8];
    #pragma unroll
    for (int re = 0; re < 8; re++) {
        float m = s[0][re];
        #pragma unroll
        for (int c = 1; c < 8; c++) m = fmaxf(m, s[c][re]);
        m = fmaxf(m, __shfl_xor(m, 1)); m = fmaxf(m, __shfl_xor(m, 2));
        m = fmaxf(m, __shfl_xor(m, 4)); m = fmaxf(m, __shfl_xor(m, 8));
        mx[re] = m; sm[re] = 0.0f;
    }
    // exp((s - max) * scale) == softmax of s*scale (scale > 0 keeps argmax)
    #pragma unroll
    for (int c = 0; c < 8; c++)
        #pragma unroll
        for (int re = 0; re < 8; re++) {
            float p = __expf((s[c][re] - mx[re]) * scale);
            s[c][re] = p; sm[re] += p;
        }
    #pragma unroll
    for (int re = 0; re < 8; re++) {
        float t = sm[re];
        t += __shfl_xor(t, 1); t += __shfl_xor(t, 2);
        t += __shfl_xor(t, 4); t += __shfl_xor(t, 8);
        inv[re] = 1.0f / t;
    }
    #pragma unroll
    for (int re = 0; re < 8; re++) {
        int R = wave * 16 + half * 8 + re;
        #pragma unroll
        for (int c = 0; c < 8; c++)
            Psh[R * 136 + c * 16 + l16] = f2bf(s[c][re]);
    }
    __syncthreads();

    int Hsp = branch == 0 ? 32 : 4;
    int Wsp = branch == 0 ? 4 : 32;
    const float* cw = branch == 0 ? cw0 : cw1;
    const float* cbias = branch == 0 ? cb0 : cb1;

    #pragma unroll
    for (int nt = 0; nt < 2; nt++) {
        v8f o = {};
        #pragma unroll
        for (int ks = 0; ks < 4; ks++) {           // PV: K = 128 tokens
            BFrag pa, vb;
            const unsigned short* prow = Psh + (wave * 16 + l16) * 136 + ks * 32;
            pa.q[0] = *(const int4*)(prow + half * 16);
            pa.q[1] = *(const int4*)(prow + half * 16 + 8);
            const int4* vp = (const int4*)(Vt + (nt * 16 + l16) * 136
                                              + ks * 32 + half * 16);
            vb.q[0] = vp[0]; vb.q[1] = vp[1];
            o = __builtin_amdgcn_wmma_f32_16x16x32_bf16(
                false, pa.v, false, vb.v, (short)0, o, false, false);
        }
        int d = nt * 16 + l16;                     // output head-dim (= D col)
        int cc = h4 * 32 + d;                      // channel within 128-ch branch
        float w9[9], bias = cbias[cc];
        #pragma unroll
        for (int j = 0; j < 9; j++) w9[j] = cw[cc * 9 + j];
        #pragma unroll
        for (int re = 0; re < 8; re++) {
            int T = wave * 16 + half * 8 + re;     // token (= D row)
            int hs = branch == 0 ? (T >> 2) : (T >> 5);
            int wsp = branch == 0 ? (T & 3) : (T & 31);
            float lep = bias;                      // LEPE depthwise 3x3, SAME
            #pragma unroll
            for (int dy = -1; dy <= 1; dy++)
                #pragma unroll
                for (int dx = -1; dx <= 1; dx++) {
                    int hh = hs + dy, ww = wsp + dx;
                    if (hh >= 0 && hh < Hsp && ww >= 0 && ww < Wsp) {
                        int t2 = hh * Wsp + ww;
                        lep += bf2f(Vt[d * 136 + t2]) * w9[(dy + 1) * 3 + (dx + 1)];
                    }
                }
            float val = o[re] * inv[re] + lep;
            int n = tok2n(T);
            y[((size_t)(b * 1024 + n)) * 256 + cb + d] = f2bf(val);
        }
    }
}

// ---------------------------------------------------------------------------
// K4: out = y @ Wp^T + proj_b   (M=131072, N=256, K=256), f32 out.
// Shuffle already folded into Wp by prep_weights.
// ---------------------------------------------------------------------------
__global__ __launch_bounds__(256) void proj_gemm(
        const unsigned short* __restrict__ y, const unsigned short* __restrict__ wp,
        const float* __restrict__ bias, float* __restrict__ out) {
    int wave = threadIdx.x >> 5, lane = threadIdx.x & 31;
    int half = lane >> 4, l16 = lane & 15;
    int M0 = blockIdx.x * 128, N0 = blockIdx.y * 64;

    int arow = M0 + wave * 16 + l16;
    const unsigned short* yrow = y + (size_t)arow * 256;

    v8f acc[4] = {};
    #pragma unroll
    for (int ks = 0; ks < 8; ks++) {
        int k0 = ks * 32;
        if (ks < 7) __builtin_prefetch(yrow + k0 + 32, 0, 0);
        BFrag a;
        a.q[0] = *(const int4*)(yrow + k0 + half * 16);
        a.q[1] = *(const int4*)(yrow + k0 + half * 16 + 8);
        #pragma unroll
        for (int nt = 0; nt < 4; nt++) {
            BFrag bw;
            const int4* bp = (const int4*)(wp + (size_t)(N0 + nt * 16 + l16) * 256
                                              + k0 + half * 16);
            bw.q[0] = bp[0]; bw.q[1] = bp[1];
            acc[nt] = __builtin_amdgcn_wmma_f32_16x16x32_bf16(
                false, a.v, false, bw.v, (short)0, acc[nt], false, false);
        }
    }
    #pragma unroll
    for (int nt = 0; nt < 4; nt++) {
        int c_out = N0 + nt * 16 + l16;
        float bb = bias[c_out];
        #pragma unroll
        for (int re = 0; re < 8; re++) {
            int r_out = M0 + wave * 16 + half * 8 + re;
            out[(size_t)r_out * 256 + c_out] = acc[nt][re] + bb;
        }
    }
}

// ---------------------------------------------------------------------------
extern "C" void kernel_launch(void* const* d_in, const int* in_sizes, int n_in,
                              void* d_out, int out_size, void* d_ws, size_t ws_size,
                              hipStream_t stream) {
    const float* x    = (const float*)d_in[0];
    const float* qk_w = (const float*)d_in[1];
    const float* v_w  = (const float*)d_in[2];
    const float* cw0  = (const float*)d_in[3];
    const float* cb0  = (const float*)d_in[4];
    const float* cw1  = (const float*)d_in[5];
    const float* cb1  = (const float*)d_in[6];
    const float* pw   = (const float*)d_in[7];
    const float* pb   = (const float*)d_in[8];

    // workspace layout (bf16/ushort elements): qkv | y | wqkv | wp
    unsigned short* ws   = (unsigned short*)d_ws;
    unsigned short* qkv  = ws;                                   // 131072*768
    unsigned short* ybuf = qkv + (size_t)131072 * 768;           // 131072*256
    unsigned short* wqkv = ybuf + (size_t)131072 * 256;          // 768*256
    unsigned short* wp   = wqkv + 768 * 256;                     // 256*256

    prep_weights<<<1024, 256, 0, stream>>>(qk_w, v_w, pw, wqkv, wp);
    qkv_gemm<<<dim3(1024, 12), 256, 0, stream>>>(x, wqkv, qkv);
    attn_kernel<<<8192, 256, 0, stream>>>(qkv, cw0, cb0, cw1, cb1, ybuf);
    proj_gemm<<<dim3(1024, 4), 256, 0, stream>>>(ybuf, wp, pb, (float*)d_out);
}
